// SRNO_38637525795035
// MI455X (gfx1250) — compile-verified, matrix operand order checked by reading, not run on hardware
//
#include <hip/hip_runtime.h>
#include <math.h>

typedef _Float16 f16;
typedef _Float16 v16h __attribute__((ext_vector_type(16)));
typedef _Float16 v8h  __attribute__((ext_vector_type(8)));
typedef float    v8f  __attribute__((ext_vector_type(8)));
typedef float    v2f  __attribute__((ext_vector_type(2)));

__device__ __forceinline__ float gelu_exact(float x) {
  return 0.5f * x * (1.0f + erff(x * 0.70710678118654752f));
}

__device__ __forceinline__ v16h make_frag(v8h lo, v8h hi) {
  v16h r;
#pragma unroll
  for (int i = 0; i < 8; i++) { r[i] = lo[i]; r[i + 8] = hi[i]; }
  return r;
}

// ---------------------------------------------------------------- utilities
__global__ void pack_f16(const float* __restrict__ src, f16* __restrict__ dst,
                         int O, int Kin, int Kpad) {
  int i = blockIdx.x * blockDim.x + threadIdx.x;
  if (i >= O * Kpad) return;
  int o = i / Kpad, k = i - o * Kpad;
  dst[i] = (f16)((k < Kin) ? src[o * Kin + k] : 0.0f);
}

__global__ void zero_f32(float* __restrict__ p, int n) {
  int i = blockIdx.x * blockDim.x + threadIdx.x;
  if (i < n) p[i] = 0.0f;
}

// ------------------------------------------------------------- grid builder
__global__ void build_grid(const float* __restrict__ feat,   // (64,64,64)
                           const float* __restrict__ coord,  // (65536,2)
                           const float* __restrict__ cell,   // (2)
                           f16* __restrict__ G) {            // (65536,288)
  int n = blockIdx.x * blockDim.x + threadIdx.x;
  if (n >= 65536) return;
  const float cy0 = coord[2 * n + 0], cx0 = coord[2 * n + 1];
  const float r = 1.0f / 64.0f;
  int iy[4], ix[4];
  float area[4];
  f16* g = G + (size_t)n * 288;
  int s = 0;
#pragma unroll
  for (int a = 0; a < 2; a++) {
    float vx = a ? 1.0f : -1.0f;
#pragma unroll
    for (int b = 0; b < 2; b++) {
      float vy = b ? 1.0f : -1.0f;
      float cy = fminf(fmaxf(cy0 + vx * r + 1e-6f, -1.0f + 1e-6f), 1.0f - 1e-6f);
      float cx = fminf(fmaxf(cx0 + vy * r + 1e-6f, -1.0f + 1e-6f), 1.0f - 1e-6f);
      int yy = (int)roundf((cy + 1.0f) * 32.0f - 0.5f); yy = min(max(yy, 0), 63);
      int xx = (int)roundf((cx + 1.0f) * 32.0f - 0.5f); xx = min(max(xx, 0), 63);
      float oy = -1.0f + 1.0f / 64.0f + (2.0f / 64.0f) * (float)yy;
      float ox = -1.0f + 1.0f / 64.0f + (2.0f / 64.0f) * (float)xx;
      float ry  = (cy0 - oy) * 64.0f;
      float rxv = (cx0 - ox) * 64.0f;
      area[s] = fabsf(ry * rxv) + 1e-9f;
      iy[s] = yy; ix[s] = xx;
      g[2 * s + 0] = (f16)ry;
      g[2 * s + 1] = (f16)rxv;
      s++;
    }
  }
  float tot = area[0] + area[1] + area[2] + area[3];
#pragma unroll
  for (int q = 0; q < 4; q++) {
    float sc = area[3 - q] / tot;                  // diagonal swap
    const float* fp = feat + iy[q] * 64 + ix[q];
    for (int c = 0; c < 64; c++)
      g[8 + q * 64 + c] = (f16)(fp[c * 4096] * sc);
  }
  g[264] = (f16)(cell[0] * 64.0f);
  g[265] = (f16)(cell[1] * 64.0f);
#pragma unroll
  for (int c = 266; c < 288; c++) g[c] = (f16)0.0f;
}

// --------------------------------------------------------------- WMMA GEMM
// Out(N,O) = A(N,K) @ W(O,K)^T + bias, optional gelu/residual, f32/f16 outs.
// Block: 256 thr = 8 waves; wave tile 16(M)x64(O); block tile 64(M)x128(O).
// Double-buffered LDS A tile: one barrier per K-chunk, next global A load
// overlaps the 4 batched WMMAs.
__launch_bounds__(256)
__global__ void gemm_bias(const f16* __restrict__ A, const f16* __restrict__ W,
                          const float* __restrict__ bias,
                          float* __restrict__ outF32, f16* __restrict__ outF16,
                          const float* __restrict__ resid,
                          int K, int O, int doGelu) {
  __shared__ f16 As[2][64 * 32];
  const int tid = threadIdx.x;
  const int lane = tid & 31, wid = tid >> 5;
  const int hi = lane >> 4, lo = lane & 15;
  const int m0 = blockIdx.x * 64;
  const int o0 = blockIdx.y * 128;
  const int mt = wid & 3, wc = wid >> 2;
  const int mrow = m0 + mt * 16;
  const int ocol = o0 + wc * 64;

  v8f acc[4] = {{}, {}, {}, {}};

  const int arow = tid >> 2;        // 0..63
  const int acol = (tid & 3) * 8;   // 0,8,16,24
  const f16* aptr = &A[(size_t)(m0 + arow) * K + acol];

  v8h pre = *(const v8h*)aptr;      // preload chunk 0
  int buf = 0;

  for (int k0 = 0; k0 < K; k0 += 32) {
    *(v8h*)&As[buf][arow * 32 + acol] = pre;
    __syncthreads();
    if (k0 + 32 < K) {
      pre = *(const v8h*)(aptr + k0 + 32);
      if (k0 + 64 < K)
        __builtin_prefetch(aptr + k0 + 64, 0, 1);
    }
    v16h af = make_frag(*(const v8h*)&As[buf][(mt * 16 + lo) * 32 + hi * 8],
                        *(const v8h*)&As[buf][(mt * 16 + lo) * 32 + 16 + hi * 8]);
    v16h bfr[4];
#pragma unroll
    for (int nt = 0; nt < 4; nt++) {
      const f16* wr = W + (size_t)(ocol + nt * 16 + lo) * K + k0;
      bfr[nt] = make_frag(*(const v8h*)&wr[hi * 8], *(const v8h*)&wr[16 + hi * 8]);
    }
#pragma unroll
    for (int nt = 0; nt < 4; nt++)
      acc[nt] = __builtin_amdgcn_wmma_f32_16x16x32_f16(
          false, af, false, bfr[nt], (short)0, acc[nt], false, false);
    buf ^= 1;
  }

#pragma unroll
  for (int nt = 0; nt < 4; nt++) {
    int oc = ocol + nt * 16 + lo;
    float b = bias ? bias[oc] : 0.0f;
#pragma unroll
    for (int rr = 0; rr < 8; rr++) {
      int m = mrow + hi * 8 + rr;
      float v = acc[nt][rr] + b;
      if (resid)  v += resid[(size_t)m * O + oc];
      if (doGelu) v = gelu_exact(v);
      if (outF32) outF32[(size_t)m * O + oc] = v;
      if (outF16) outF16[(size_t)m * O + oc] = (f16)v;
    }
  }
}

// ------------------------------------------------- fused QKV + LN + kv-accum
// Wave = one head x 16-row tile (loops 16 tiles = 256 rows per block).
// Epilogue: q->Q32; LN(k),LN(v) via 16-lane shuffles; kv partials -> atomics.
__launch_bounds__(256)
__global__ void qkv_fused(const f16* __restrict__ Xf16, const f16* __restrict__ W,
                          const float* __restrict__ bias,
                          const float* __restrict__ klw, const float* __restrict__ klb,
                          const float* __restrict__ vlw, const float* __restrict__ vlb,
                          float* __restrict__ Q32, float* __restrict__ kvbuf) {
  __shared__ f16 As[2][16 * 256];
  const int tid = threadIdx.x;
  const int lane = tid & 31, wid = tid >> 5;
  const int hi = lane >> 4, lo = lane & 15;
  const int rg = blockIdx.x >> 1;
  const int hg = blockIdx.x & 1;
  const int h  = hg * 8 + wid;
  const int m0 = rg * 256;

  const float kw = klw[h * 16 + lo], kb = klb[h * 16 + lo];
  const float vw = vlw[h * 16 + lo], vb = vlb[h * 16 + lo];
  const float bq = bias[h * 48 + lo];
  const float bk = bias[h * 48 + 16 + lo];
  const float bv = bias[h * 48 + 32 + lo];

  float kvacc[16];
#pragma unroll
  for (int d = 0; d < 16; d++) kvacc[d] = 0.0f;

  const int lrow = tid >> 4;
  const int lcol = (tid & 15) * 16;
  const f16* xptr = &Xf16[(size_t)(m0 + lrow) * 256 + lcol];

  v8h pre0 = *(const v8h*)xptr;
  v8h pre1 = *(const v8h*)(xptr + 8);
  int buf = 0;

  for (int rt = 0; rt < 16; rt++) {
    const int mbase = m0 + rt * 16;
    *(v8h*)&As[buf][lrow * 256 + lcol]     = pre0;
    *(v8h*)&As[buf][lrow * 256 + lcol + 8] = pre1;
    __syncthreads();
    if (rt < 15) {
      const f16* nx = xptr + (size_t)(rt + 1) * 16 * 256;
      pre0 = *(const v8h*)nx;
      pre1 = *(const v8h*)(nx + 8);
    }

    v8f aq = {}, ak = {}, av = {};
#pragma unroll
    for (int kc = 0; kc < 8; kc++) {
      const int k0 = kc * 32;
      v16h af = make_frag(*(const v8h*)&As[buf][lo * 256 + k0 + hi * 8],
                          *(const v8h*)&As[buf][lo * 256 + k0 + 16 + hi * 8]);
      const f16* wq = W + (size_t)(h * 48 + lo) * 256 + k0;
      const f16* wk = W + (size_t)(h * 48 + 16 + lo) * 256 + k0;
      const f16* wv = W + (size_t)(h * 48 + 32 + lo) * 256 + k0;
      v16h bqf = make_frag(*(const v8h*)&wq[hi * 8], *(const v8h*)&wq[16 + hi * 8]);
      v16h bkf = make_frag(*(const v8h*)&wk[hi * 8], *(const v8h*)&wk[16 + hi * 8]);
      v16h bvf = make_frag(*(const v8h*)&wv[hi * 8], *(const v8h*)&wv[16 + hi * 8]);
      aq = __builtin_amdgcn_wmma_f32_16x16x32_f16(false, af, false, bqf, (short)0, aq, false, false);
      ak = __builtin_amdgcn_wmma_f32_16x16x32_f16(false, af, false, bkf, (short)0, ak, false, false);
      av = __builtin_amdgcn_wmma_f32_16x16x32_f16(false, af, false, bvf, (short)0, av, false, false);
    }

#pragma unroll
    for (int rr = 0; rr < 8; rr++) {
      const int m = mbase + hi * 8 + rr;
      Q32[(size_t)m * 256 + h * 16 + lo] = aq[rr] + bq;

      // LN(k): mean over 16 channels (16-lane half-wave), unbiased std (n-1).
      float xk = ak[rr] + bk;
      float s = xk;
      s += __shfl_xor(s, 1); s += __shfl_xor(s, 2);
      s += __shfl_xor(s, 4); s += __shfl_xor(s, 8);
      float ck = xk - s * 0.0625f;
      float ss = ck * ck;
      ss += __shfl_xor(ss, 1); ss += __shfl_xor(ss, 2);
      ss += __shfl_xor(ss, 4); ss += __shfl_xor(ss, 8);
      float kn = kw * (ck / (sqrtf(ss * (1.0f / 15.0f)) + 1e-5f)) + kb;

      float xv = av[rr] + bv;
      float s2 = xv;
      s2 += __shfl_xor(s2, 1); s2 += __shfl_xor(s2, 2);
      s2 += __shfl_xor(s2, 4); s2 += __shfl_xor(s2, 8);
      float cv = xv - s2 * 0.0625f;
      float ss2 = cv * cv;
      ss2 += __shfl_xor(ss2, 1); ss2 += __shfl_xor(ss2, 2);
      ss2 += __shfl_xor(ss2, 4); ss2 += __shfl_xor(ss2, 8);
      float vn = vw * (cv / (sqrtf(ss2 * (1.0f / 15.0f)) + 1e-5f)) + vb;

      const int base = hi << 4;
#pragma unroll
      for (int d = 0; d < 16; d++) {
        float vd = __shfl(vn, base + d);
        kvacc[d] += kn * vd;      // kv[c=lo][d] partial over this wave's rows
      }
    }
    buf ^= 1;
  }

#pragma unroll
  for (int d = 0; d < 16; d++) kvacc[d] += __shfl_xor(kvacc[d], 16);
  if (lane < 16) {
#pragma unroll
    for (int d = 0; d < 16; d++)
      atomicAdd(&kvbuf[(h * 16 + lo) * 16 + d], kvacc[d]);
  }
}

// ------------------------------------------------- o = q @ kv (f32 WMMA) + x
__launch_bounds__(256)
__global__ void attn_o(const float* __restrict__ Q32, const float* __restrict__ kvbuf,
                       const float* __restrict__ X32,
                       float* __restrict__ Ret32, f16* __restrict__ Ret16) {
  const int tid = threadIdx.x;
  const int lane = tid & 31, wid = tid >> 5;
  const int hi = lane >> 4, lo = lane & 15;
  const int rt = blockIdx.x >> 1;
  const int hg = blockIdx.x & 1;
  const int h = hg * 8 + wid;
  const int m0 = rt * 16;
  const float scale = 1.0f / 65536.0f;   // / (H*W)

  v8f c = {};
#if __has_builtin(__builtin_amdgcn_wmma_f32_16x16x4_f32)
#pragma unroll
  for (int kc = 0; kc < 4; kc++) {
    const int k = kc * 4 + hi * 2;
    v2f a, b;
    a.x = Q32[(size_t)(m0 + lo) * 256 + h * 16 + k];
    a.y = Q32[(size_t)(m0 + lo) * 256 + h * 16 + k + 1];
    b.x = kvbuf[(h * 16 + k) * 16 + lo] * scale;
    b.y = kvbuf[(h * 16 + k + 1) * 16 + lo] * scale;
    c = __builtin_amdgcn_wmma_f32_16x16x4_f32(false, a, false, b, (short)0, c, false, false);
  }
#else
#pragma unroll
  for (int rr = 0; rr < 8; rr++) {
    const int m = m0 + hi * 8 + rr;
    float accv = 0.0f;
    for (int k = 0; k < 16; k++)
      accv += Q32[(size_t)m * 256 + h * 16 + k] * kvbuf[(h * 16 + k) * 16 + lo] * scale;
    c[rr] = accv;
  }
#endif
#pragma unroll
  for (int rr = 0; rr < 8; rr++) {
    const int m = m0 + hi * 8 + rr;
    const int oc = h * 16 + lo;
    float v = c[rr] + X32[(size_t)m * 256 + oc];
    Ret32[(size_t)m * 256 + oc] = v;
    Ret16[(size_t)m * 256 + oc] = (f16)v;
  }
}

// --------------------------------------------- fc2 + bilinear-border + write
__global__ void final_fc2(const f16* __restrict__ H, const float* __restrict__ w2,
                          const float* __restrict__ b2,
                          const float* __restrict__ inp,    // (3,64,64)
                          const float* __restrict__ coord,
                          float* __restrict__ out) {        // (3,65536)
  int n = blockIdx.x * blockDim.x + threadIdx.x;
  if (n >= 65536) return;
  const f16* hr = H + (size_t)n * 256;
  float a0 = b2[0], a1 = b2[1], a2 = b2[2];
  for (int cc = 0; cc < 256; cc++) {
    float hv = (float)hr[cc];
    a0 += w2[cc] * hv; a1 += w2[256 + cc] * hv; a2 += w2[512 + cc] * hv;
  }
  float cy = coord[2 * n], cx = coord[2 * n + 1];
  float fy = (cy + 1.0f) * 32.0f - 0.5f;
  float fx = (cx + 1.0f) * 32.0f - 0.5f;
  float y0 = floorf(fy), x0 = floorf(fx);
  float wy = fy - y0, wx = fx - x0;
  int y0i = min(max((int)y0, 0), 63), y1i = min(max((int)y0 + 1, 0), 63);
  int x0i = min(max((int)x0, 0), 63), x1i = min(max((int)x0 + 1, 0), 63);
  float w00 = (1 - wy) * (1 - wx), w01 = (1 - wy) * wx;
  float w10 = wy * (1 - wx),       w11 = wy * wx;
  float acc[3] = {a0, a1, a2};
#pragma unroll
  for (int cch = 0; cch < 3; cch++) {
    const float* ic = inp + cch * 4096;
    float bl = ic[y0i * 64 + x0i] * w00 + ic[y0i * 64 + x1i] * w01 +
               ic[y1i * 64 + x0i] * w10 + ic[y1i * 64 + x1i] * w11;
    out[cch * 65536 + n] = acc[cch] + bl;
  }
}

// ---------------------------------------------------------------- launcher
extern "C" void kernel_launch(void* const* d_in, const int* in_sizes, int n_in,
                              void* d_out, int out_size, void* d_ws, size_t ws_size,
                              hipStream_t stream) {
  const float* feat  = (const float*)d_in[0];
  const float* inp   = (const float*)d_in[1];
  const float* coord = (const float*)d_in[2];
  const float* cell  = (const float*)d_in[3];
  const float* c00w  = (const float*)d_in[4];
  const float* c00b  = (const float*)d_in[5];
  const float* qkvw  = (const float*)d_in[6];
  const float* qkvb  = (const float*)d_in[7];
  const float* o1w   = (const float*)d_in[8];
  const float* o1b   = (const float*)d_in[9];
  const float* o2w   = (const float*)d_in[10];
  const float* o2b   = (const float*)d_in[11];
  const float* klnw  = (const float*)d_in[12];
  const float* klnb  = (const float*)d_in[13];
  const float* vlnw  = (const float*)d_in[14];
  const float* vlnb  = (const float*)d_in[15];
  const float* fc1w  = (const float*)d_in[16];
  const float* fc1b  = (const float*)d_in[17];
  const float* fc2w  = (const float*)d_in[18];
  const float* fc2b  = (const float*)d_in[19];
  float* out = (float*)d_out;
  (void)in_sizes; (void)n_in; (void)out_size; (void)ws_size;

  char* ws = (char*)d_ws;
  size_t off = 0;
  auto alloc = [&](size_t bytes) -> char* {
    char* p = ws + off;
    off = (off + bytes + 255) & ~(size_t)255;
    return p;
  };
  const size_t N = 65536;
  f16*   Wc00 = (f16*)alloc(256 * 288 * sizeof(f16));
  f16*   Wqkv = (f16*)alloc(2 * 768 * 256 * sizeof(f16));
  f16*   Wo1  = (f16*)alloc(2 * 256 * 256 * sizeof(f16));
  f16*   Wo2  = (f16*)alloc(2 * 256 * 256 * sizeof(f16));
  f16*   Wfc1 = (f16*)alloc(256 * 256 * sizeof(f16));
  f16*   G    = (f16*)alloc(N * 288 * sizeof(f16));
  float* X32  = (float*)alloc(N * 256 * sizeof(float));
  f16*   X16  = (f16*)alloc(N * 256 * sizeof(f16));
  float* Q32  = (float*)alloc(N * 256 * sizeof(float));
  float* R32  = (float*)alloc(N * 256 * sizeof(float));
  f16*   R16  = (f16*)alloc(N * 256 * sizeof(f16));
  f16*   T16  = (f16*)alloc(N * 256 * sizeof(f16));
  float* KV   = (float*)alloc(4096 * sizeof(float));

  auto packs = [&](const float* s, f16* dptr, int O, int Kin, int Kpad) {
    int tot = O * Kpad;
    pack_f16<<<(tot + 255) / 256, 256, 0, stream>>>(s, dptr, O, Kin, Kpad);
  };
  packs(c00w, Wc00, 256, 266, 288);
  packs(qkvw, Wqkv, 1536, 256, 256);
  packs(o1w,  Wo1,  512, 256, 256);
  packs(o2w,  Wo2,  512, 256, 256);
  packs(fc1w, Wfc1, 256, 256, 256);

  for (int p = 0; p < 2; p++) {
    const float* featp = feat + (size_t)p * 64 * 64 * 64;
    const float* inpp  = inp + (size_t)p * 3 * 64 * 64;

    build_grid<<<N / 256, 256, 0, stream>>>(featp, coord, cell, G);

    gemm_bias<<<dim3(N / 64, 2), 256, 0, stream>>>(
        G, Wc00, c00b, X32, X16, nullptr, 288, 256, 0);

    for (int L = 0; L < 2; L++) {
      zero_f32<<<16, 256, 0, stream>>>(KV, 4096);
      qkv_fused<<<(N / 256) * 2, 256, 0, stream>>>(
          X16, Wqkv + (size_t)L * 768 * 256, qkvb + L * 768,
          klnw + L * 256, klnb + L * 256, vlnw + L * 256, vlnb + L * 256,
          Q32, KV);
      attn_o<<<(N / 16) * 2, 256, 0, stream>>>(Q32, KV, X32, R32, R16);
      gemm_bias<<<dim3(N / 64, 2), 256, 0, stream>>>(
          R16, Wo1 + (size_t)L * 256 * 256, o1b + L * 256,
          nullptr, T16, nullptr, 256, 256, 1);
      gemm_bias<<<dim3(N / 64, 2), 256, 0, stream>>>(
          T16, Wo2 + (size_t)L * 256 * 256, o2b + L * 256,
          X32, X16, X32, 256, 256, 0);
    }

    gemm_bias<<<dim3(N / 64, 2), 256, 0, stream>>>(
        X16, Wfc1, fc1b, nullptr, T16, nullptr, 256, 256, 1);

    final_fc2<<<N / 256, 256, 0, stream>>>(
        T16, fc2w, fc2b, inpp, coord, out + (size_t)p * 3 * 65536);
  }
}
